// MultiHeadAttention_4595615006930
// MI455X (gfx1250) — matrix operand order repaired and checked by reading, not compiled, not run on hardware
//
#include <hip/hip_runtime.h>

// ---------------------------------------------------------------------------
// MHA forward for MI455X (gfx1250, wave32, WMMA).
//   B=2, S=4096, D=768, H=12, dh=64
// Pass 1: Q/K/V projections via v_wmma_f32_16x16x32_bf16, bf16 head-major out.
// Pass 2: flash attention (online softmax), QK^T and PV via bf16 WMMA.
// WMMA operands stream from LDS as 16B-aligned b128 chunks; all B-fragments
// are preloaded before each WMMA chain so the matrix ops issue back-to-back.
// ---------------------------------------------------------------------------

#define BATCH 2
#define SEQ   4096
#define DMODEL 768
#define NHEAD 12
#define DHEAD 64
#define ROWS  (BATCH * SEQ)            // 8192
#define HEAD_ELEMS ((size_t)BATCH * NHEAD * SEQ * DHEAD)  // 6291456 per tensor

typedef __attribute__((ext_vector_type(16))) __bf16 v16bf;
typedef __attribute__((ext_vector_type(4)))  __bf16 v4bf;
typedef __attribute__((ext_vector_type(8)))  float  v8f;
typedef __attribute__((ext_vector_type(4)))  unsigned int   v4u;

#define WMMA_BF16(A, Bm, C) \
  __builtin_amdgcn_wmma_f32_16x16x32_bf16(false, (A), false, (Bm), (short)0, (C), false, false)

// Native f32 -> bf16 convert (hardware cvt, RNE).
__device__ __forceinline__ unsigned short bf16_bits(float f) {
  return __builtin_bit_cast(unsigned short, (__bf16)f);
}

union FragU { v16bf bf; struct { v4u lo; v4u hi; } q; };

// Build a v16bf operand from two 16-byte chunks (elements 0..7 and 8..15).
__device__ __forceinline__ v16bf load_frag2(const unsigned short* pa,
                                            const unsigned short* pb) {
  FragU f;
  f.q.lo = *(const v4u*)pa;
  f.q.hi = *(const v4u*)pb;
  return f.bf;
}
// 16 contiguous bf16 elements.
__device__ __forceinline__ v16bf load_frag16(const unsigned short* p) {
  return load_frag2(p, p + 8);
}

// ===========================================================================
// Kernel 1: C[row, head*64+d] = X[row,:] @ W[:, head*64+d]  (bf16 out, scaled)
// Block: 256 thr (8 waves). Tile: 128 rows x 64 cols (one head), K-step 32.
// Output layout: [B, H, S, 64] bf16 (head-major for attention pass).
// ===========================================================================
__global__ __launch_bounds__(256)
void mha_proj_kernel(const float* __restrict__ X, const float* __restrict__ W,
                     unsigned short* __restrict__ out, float scale) {
  __shared__ __align__(16) unsigned short sA[128][32];   // A tile, bf16 (row-major)
  __shared__ __align__(16) unsigned short sWT[64][32];   // W tile, bf16 TRANSPOSED [n][k]

  const int tid  = threadIdx.x;
  const int wave = tid >> 5;
  const int lane = tid & 31;
  const int half = lane >> 4;
  const int l16  = lane & 15;
  const int rowBase = blockIdx.x * 128;
  const int head    = blockIdx.y;              // 64 output cols == one head

  v8f acc[4] = {v8f{}, v8f{}, v8f{}, v8f{}};

  for (int k0 = 0; k0 < DMODEL; k0 += 32) {
    // ---- A tile: 128x32 f32 -> bf16, vector loads + b64 LDS stores ---------
#pragma unroll
    for (int u = 0; u < 4; ++u) {
      int f4 = tid + u * 256;                  // 1024 float4s total
      int r = f4 >> 3, c4 = (f4 & 7) * 4;
      float4 xv = *(const float4*)(X + (size_t)(rowBase + r) * DMODEL + k0 + c4);
      v4bf p;
      p[0] = (__bf16)xv.x; p[1] = (__bf16)xv.y;
      p[2] = (__bf16)xv.z; p[3] = (__bf16)xv.w;
      *(v4bf*)&sA[r][c4] = p;
    }
    // ---- W tile: 32x64 f32 -> bf16 transposed [n][k] -----------------------
#pragma unroll
    for (int u = 0; u < 2; ++u) {
      int f4 = tid + u * 256;                  // 512 float4s total
      int r = f4 >> 4, c4 = (f4 & 15) * 4;
      float4 wv = *(const float4*)(W + (size_t)(k0 + r) * DMODEL + head * 64 + c4);
      sWT[c4 + 0][r] = bf16_bits(wv.x);
      sWT[c4 + 1][r] = bf16_bits(wv.y);
      sWT[c4 + 2][r] = bf16_bits(wv.z);
      sWT[c4 + 3][r] = bf16_bits(wv.w);
    }
    __syncthreads();

    // A-frag: one contiguous 16-elem run (K = 16*half+j)
    const unsigned short* arow = &sA[wave * 16 + l16][0];
    v16bf a = load_frag16(arow + 16 * half);

    // Preload all four B-frags, then issue the WMMA chain back-to-back.
    v16bf b0 = load_frag16(&sWT[ 0 + l16][16 * half]);
    v16bf b1 = load_frag16(&sWT[16 + l16][16 * half]);
    v16bf b2 = load_frag16(&sWT[32 + l16][16 * half]);
    v16bf b3 = load_frag16(&sWT[48 + l16][16 * half]);
    acc[0] = WMMA_BF16(a, b0, acc[0]);
    acc[1] = WMMA_BF16(a, b1, acc[1]);
    acc[2] = WMMA_BF16(a, b2, acc[2]);
    acc[3] = WMMA_BF16(a, b3, acc[3]);
    __syncthreads();
  }

  // Epilogue: one base pointer, constant immediate offsets per store.
  // A 128-row tile never straddles the batch boundary.
  const int rw  = rowBase + wave * 16;
  const int bbk = rw >> 12;
  const int ssb = (rw & (SEQ - 1)) + 8 * half;        // C-layout: M = e + 8*half
  unsigned short* outp =
      out + (((size_t)bbk * NHEAD + head) * SEQ + ssb) * DHEAD + l16;
#pragma unroll
  for (int t = 0; t < 4; ++t)
#pragma unroll
    for (int e = 0; e < 8; ++e)
      outp[e * DHEAD + t * 16] = bf16_bits(acc[t][e] * scale);
}

// ===========================================================================
// Kernel 2: flash attention. Grid (S/128, H, B), block 256 (8 waves).
// Each wave owns 16 q-rows; K (row-major) and V (transposed) tiles in LDS.
// ===========================================================================
__global__ __launch_bounds__(256)
void mha_flash_kernel(const unsigned short* __restrict__ Qh,
                      const unsigned short* __restrict__ Kh,
                      const unsigned short* __restrict__ Vh,
                      float* __restrict__ out) {
  __shared__ __align__(16) unsigned short sK [32][64];       // [key][d]
  __shared__ __align__(16) unsigned short sVT[64][32];       // [d][key] transposed
  __shared__ __align__(16) unsigned short sP [8][16][32];    // per-wave P staging

  const int tid  = threadIdx.x;
  const int wave = tid >> 5;
  const int lane = tid & 31;
  const int half = lane >> 4;
  const int l16  = lane & 15;
  const int bb = blockIdx.z;
  const int hh = blockIdx.y;
  const int qBase = blockIdx.x * 128 + wave * 16;
  const size_t headOff = ((size_t)bb * NHEAD + hh) * SEQ;

  // Q fragments held in registers for the whole kv sweep (d 0..31 and 32..63),
  // loaded as four global b128s per lane.
  const unsigned short* qrow = Qh + (headOff + qBase + l16) * DHEAD;
  v16bf aq0 = load_frag16(qrow +      16 * half);
  v16bf aq1 = load_frag16(qrow + 32 + 16 * half);

  v8f oacc[4] = {v8f{}, v8f{}, v8f{}, v8f{}};
  float mrow[8], lrow[8];
#pragma unroll
  for (int e = 0; e < 8; ++e) { mrow[e] = -1e30f; lrow[e] = 0.0f; }

  for (int k0 = 0; k0 < SEQ; k0 += 32) {
    const size_t kvoff = (headOff + k0) * DHEAD;

    // ---- cooperative fill: K row-major (b128), V transposed (scatter) ------
    {
      int r = tid >> 3, c8 = (tid & 7) * 8;          // 256 threads x 8 elems
      v4u kv = *(const v4u*)(Kh + kvoff + (size_t)r * DHEAD + c8);
      *(v4u*)&sK[r][c8] = kv;
      union { v4u v; unsigned short u[8]; } tv;
      tv.v = *(const v4u*)(Vh + kvoff + (size_t)r * DHEAD + c8);
#pragma unroll
      for (int j = 0; j < 8; ++j) sVT[c8 + j][r] = tv.u[j];
    }
    if (k0 + 32 < SEQ) {                              // prefetch next tile
      __builtin_prefetch(Kh + kvoff + 32 * DHEAD, 0, 0);
      __builtin_prefetch(Vh + kvoff + 32 * DHEAD, 0, 0);
    }
    __syncthreads();

    // ---- scores: Sc = Q (16x64) x K^T (64x32), two 16-key tiles ------------
    // Preload all four K-frags, then 4 WMMAs back-to-back.
    v16bf bk00 = load_frag16(&sK[l16]     [     16 * half]);
    v16bf bk01 = load_frag16(&sK[l16]     [32 + 16 * half]);
    v16bf bk10 = load_frag16(&sK[16 + l16][     16 * half]);
    v16bf bk11 = load_frag16(&sK[16 + l16][32 + 16 * half]);
    v8f sc0 = {}, sc1 = {};
    sc0 = WMMA_BF16(aq0, bk00, sc0);
    sc0 = WMMA_BF16(aq1, bk01, sc0);
    sc1 = WMMA_BF16(aq0, bk10, sc1);
    sc1 = WMMA_BF16(aq1, bk11, sc1);

    // ---- online softmax (row stats across the 16 lanes of each half) -------
#pragma unroll
    for (int e = 0; e < 8; ++e) {
      float mloc = fmaxf(sc0[e], sc1[e]);
      for (int mask = 1; mask < 16; mask <<= 1)
        mloc = fmaxf(mloc, __shfl_xor(mloc, mask, 32));
      float mnew  = fmaxf(mrow[e], mloc);
      float alpha = __expf(mrow[e] - mnew);
      float p0 = __expf(sc0[e] - mnew);
      float p1 = __expf(sc1[e] - mnew);
      float srow = p0 + p1;
      for (int mask = 1; mask < 16; mask <<= 1)
        srow += __shfl_xor(srow, mask, 32);
      lrow[e] = lrow[e] * alpha + srow;
      mrow[e] = mnew;
      oacc[0][e] *= alpha; oacc[1][e] *= alpha;
      oacc[2][e] *= alpha; oacc[3][e] *= alpha;
      int prow = e + 8 * half;                       // C-layout row
      sP[wave][prow][l16]      = bf16_bits(p0);
      sP[wave][prow][16 + l16] = bf16_bits(p1);
    }
    // same-wave LDS writes/reads are in-order: no block barrier needed for sP

    // ---- O += P (16x32) x V (32x64), V read from transposed tile -----------
    const unsigned short* prow_p = &sP[wave][l16][0];
    v16bf ap  = load_frag16(prow_p + 16 * half);
    v16bf bv0 = load_frag16(&sVT[ 0 + l16][16 * half]);
    v16bf bv1 = load_frag16(&sVT[16 + l16][16 * half]);
    v16bf bv2 = load_frag16(&sVT[32 + l16][16 * half]);
    v16bf bv3 = load_frag16(&sVT[48 + l16][16 * half]);
    oacc[0] = WMMA_BF16(ap, bv0, oacc[0]);
    oacc[1] = WMMA_BF16(ap, bv1, oacc[1]);
    oacc[2] = WMMA_BF16(ap, bv2, oacc[2]);
    oacc[3] = WMMA_BF16(ap, bv3, oacc[3]);
    __syncthreads();
  }

  // ---- epilogue: normalize, one base pointer + immediate-offset stores -----
#pragma unroll
  for (int e = 0; e < 8; ++e) lrow[e] = 1.0f / lrow[e];
  float* outp = out + ((size_t)bb * SEQ + qBase + 8 * half) * DMODEL
              + hh * DHEAD + l16;
#pragma unroll
  for (int t = 0; t < 4; ++t)
#pragma unroll
    for (int e = 0; e < 8; ++e)
      outp[e * DMODEL + t * 16] = oacc[t][e] * lrow[e];
}

// ===========================================================================
extern "C" void kernel_launch(void* const* d_in, const int* in_sizes, int n_in,
                              void* d_out, int out_size, void* d_ws, size_t ws_size,
                              hipStream_t stream) {
  const float* q  = (const float*)d_in[0];
  const float* k  = (const float*)d_in[1];
  const float* v  = (const float*)d_in[2];
  const float* wq = (const float*)d_in[3];
  const float* wk = (const float*)d_in[4];
  const float* wv = (const float*)d_in[5];
  float* out = (float*)d_out;

  unsigned short* ws = (unsigned short*)d_ws;
  unsigned short* Qh = ws;                    // [B,H,S,64] bf16
  unsigned short* Kh = ws + HEAD_ELEMS;
  unsigned short* Vh = ws + 2 * HEAD_ELEMS;   // total 36 MB of workspace

  const float scaleQ = 0.125f;                // 1/sqrt(64), folded into Q

  dim3 gProj(ROWS / 128, NHEAD, 1);           // (64, 12)
  mha_proj_kernel<<<gProj, 256, 0, stream>>>(q, wq, Qh, scaleQ);
  mha_proj_kernel<<<gProj, 256, 0, stream>>>(k, wk, Kh, 1.0f);
  mha_proj_kernel<<<gProj, 256, 0, stream>>>(v, wv, Vh, 1.0f);

  dim3 gAttn(SEQ / 128, NHEAD, BATCH);        // (32, 12, 2)
  mha_flash_kernel<<<gAttn, 256, 0, stream>>>(Qh, Kh, Vh, out);
}
